// GraphNet_6485400616992
// MI455X (gfx1250) — matrix operand order repaired and checked by reading, not compiled, not run on hardware
//
#include <hip/hip_runtime.h>
#include <hip/hip_bf16.h>

#define N_NODES 150
#define P_FEAT  16
#define DE      14
#define DO      10
#define NT      5
#define BATCH   256
#define NR      (N_NODES * (N_NODES - 1))     // 22350
#define NTILE_E ((NR + 15) / 16)              // 1397
#define NTILE_N ((N_NODES + 15) / 16)         // 10
#define SPLIT   8

typedef __attribute__((ext_vector_type(16))) __bf16 v16bf;
typedef __attribute__((ext_vector_type(8)))  __bf16 v8bf;
typedef __attribute__((ext_vector_type(8)))  float  v8f;

__device__ __forceinline__ void wave_lds_sync() {
    // Order per-wave LDS stores before subsequent cross-lane LDS loads.
    __builtin_amdgcn_wave_barrier();
    asm volatile("s_wait_dscnt 0" ::: "memory");
    __builtin_amdgcn_wave_barrier();
}

__device__ __forceinline__ v16bf cat16(v8bf lo, v8bf hi) {
    return __builtin_shufflevector(lo, hi, 0,1,2,3,4,5,6,7,8,9,10,11,12,13,14,15);
}

__device__ __forceinline__ v8f wmma_bf16(v16bf a, v16bf b, v8f c) {
    return __builtin_amdgcn_wmma_f32_16x16x32_bf16(false, a, false, b, (short)0, c,
                                                   false, false);
}

__global__ void k_zero(float* __restrict__ p, int n) {
    int i = blockIdx.x * blockDim.x + threadIdx.x;
    if (i < n) p[i] = 0.f;
}

// ---------------------------------------------------------------------------
// Edge (relational) MLP: gather -> 32->50->25->14 bf16 WMMA -> scatter-add
// ---------------------------------------------------------------------------
__global__ __launch_bounds__(256) void k_edge(
    const float* __restrict__ x, const int* __restrict__ recv,
    const int* __restrict__ send,
    const float* __restrict__ w1, const float* __restrict__ b1,
    const float* __restrict__ w2, const float* __restrict__ b2,
    const float* __restrict__ w3, const float* __restrict__ b3,
    float* __restrict__ ebar)
{
    __shared__ __align__(32) __bf16 s_xT[N_NODES * 16];   // xT[node][feat]
    __shared__ __align__(32) __bf16 s_w1[64 * 32];        // [outPad64][kPad32]
    __shared__ __align__(32) __bf16 s_w2[32 * 64];        // [outPad32][kPad64]
    __shared__ __align__(32) __bf16 s_w3[16 * 32];        // [outPad16][kPad32]
    __shared__ float s_b1[64], s_b2[32], s_b3[16];
    __shared__ __align__(32) __bf16 s_h1[8][16 * 64];     // per-wave 16x64
    __shared__ __align__(32) __bf16 s_h2[8][16 * 32];     // per-wave 16x32

    const int tid = threadIdx.x;
    const int b   = blockIdx.x / SPLIT;
    const int sp  = blockIdx.x % SPLIT;

    // Stage x[b] transposed into LDS as bf16.
    for (int i = tid; i < P_FEAT * N_NODES; i += 256) {
        int p = i / N_NODES, n = i % N_NODES;
        s_xT[n * 16 + p] = (__bf16)x[(size_t)b * (P_FEAT * N_NODES) + p * N_NODES + n];
    }
    // fr1 (50,32) -> [64][32]
    for (int i = tid; i < 64 * 32; i += 256) {
        int n = i >> 5, k = i & 31;
        s_w1[i] = (__bf16)((n < 50) ? w1[n * 32 + k] : 0.f);
    }
    // fr2 (25,50) -> [32][64]
    for (int i = tid; i < 32 * 64; i += 256) {
        int n = i >> 6, k = i & 63;
        s_w2[i] = (__bf16)((n < 25 && k < 50) ? w2[n * 50 + k] : 0.f);
    }
    // fr3 (14,25) -> [16][32]
    for (int i = tid; i < 16 * 32; i += 256) {
        int n = i >> 5, k = i & 31;
        s_w3[i] = (__bf16)((n < 14 && k < 25) ? w3[n * 25 + k] : 0.f);
    }
    if (tid < 64) s_b1[tid] = (tid < 50) ? b1[tid] : 0.f;
    if (tid < 32) s_b2[tid] = (tid < 25) ? b2[tid] : 0.f;
    if (tid < 16) s_b3[tid] = (tid < 14) ? b3[tid] : 0.f;
    __syncthreads();

    const int lane = tid & 31;
    const int wv   = tid >> 5;
    const int lh   = lane >> 4;      // lane half
    const int nl   = lane & 15;      // col within tile / A-row
    const int f0   = lh * 8;         // A fragment K base (per 8-element half)
    const int kb   = lh * 16;        // unused (B fragments load at f0 too)

    // B fragments (constant over the tile loop).
    v16bf bw1[4];
#pragma unroll
    for (int q = 0; q < 4; ++q)
        bw1[q] = cat16(*(const v8bf*)&s_w1[(q * 16 + nl) * 32 + f0],
                       *(const v8bf*)&s_w1[(q * 16 + nl) * 32 + 16 + f0]);
    v16bf bw2[2][2];
#pragma unroll
    for (int q = 0; q < 2; ++q)
#pragma unroll
        for (int c = 0; c < 2; ++c)
            bw2[q][c] = cat16(*(const v8bf*)&s_w2[(q * 16 + nl) * 64 + c * 32 + f0],
                              *(const v8bf*)&s_w2[(q * 16 + nl) * 64 + c * 32 + 16 + f0]);
    v16bf bw3 = cat16(*(const v8bf*)&s_w3[nl * 32 + f0],
                      *(const v8bf*)&s_w3[nl * 32 + 16 + f0]);

    __bf16* h1 = s_h1[wv];
    __bf16* h2 = s_h2[wv];

    for (int t = sp * 8 + wv; t < NTILE_E; t += SPLIT * 8) {
        int e  = t * 16 + nl;
        int ec = (e < NR) ? e : (NR - 1);
        int ri = recv[ec], si = send[ec];
        // A: 16 edges x 32 feats ([recv feats | send feats]), ISA A-layout.
        v16bf a = cat16(*(const v8bf*)&s_xT[ri * 16 + f0],
                        *(const v8bf*)&s_xT[si * 16 + f0]);

        // Stage 1: 32 -> 50 (4 N-tiles)
        v8f acc[4];
#pragma unroll
        for (int q = 0; q < 4; ++q) { v8f z = {}; acc[q] = wmma_bf16(a, bw1[q], z); }
#pragma unroll
        for (int q = 0; q < 4; ++q) {
            float bv = s_b1[q * 16 + nl];
#pragma unroll
            for (int r = 0; r < 8; ++r) {
                float v = fmaxf(acc[q][r] + bv, 0.f);
                h1[(r + lh * 8) * 64 + q * 16 + nl] = (__bf16)v;
            }
        }
        wave_lds_sync();

        // Stage 2: 50(pad64) -> 25 (2 N-tiles, 2 K-chunks)
        v16bf a20 = cat16(*(const v8bf*)&h1[nl * 64 + f0],
                          *(const v8bf*)&h1[nl * 64 + 16 + f0]);
        v16bf a21 = cat16(*(const v8bf*)&h1[nl * 64 + 32 + f0],
                          *(const v8bf*)&h1[nl * 64 + 48 + f0]);
        v8f acc2[2];
#pragma unroll
        for (int q = 0; q < 2; ++q) {
            v8f z = {};
            v8f tmp = wmma_bf16(a20, bw2[q][0], z);
            acc2[q] = wmma_bf16(a21, bw2[q][1], tmp);
        }
#pragma unroll
        for (int q = 0; q < 2; ++q) {
            float bv = s_b2[q * 16 + nl];
#pragma unroll
            for (int r = 0; r < 8; ++r) {
                float v = fmaxf(acc2[q][r] + bv, 0.f);
                h2[(r + lh * 8) * 32 + q * 16 + nl] = (__bf16)v;
            }
        }
        wave_lds_sync();

        // Stage 3: 25(pad32) -> 14
        v16bf a3 = cat16(*(const v8bf*)&h2[nl * 32 + f0],
                         *(const v8bf*)&h2[nl * 32 + 16 + f0]);
        v8f z3 = {};
        v8f e3 = wmma_bf16(a3, bw3, z3);

        float bv3 = s_b3[nl];
        if (nl < DE) {
#pragma unroll
            for (int r = 0; r < 8; ++r) {
                int ee = t * 16 + r + lh * 8;
                if (ee < NR) {
                    int   rn = recv[ee];
                    float v  = fmaxf(e3[r] + bv3, 0.f);
                    atomicAdd(&ebar[((size_t)b * N_NODES + rn) * DE + nl], v);
                }
            }
        }
        wave_lds_sync();
    }
}

// ---------------------------------------------------------------------------
// Node (object) MLP: [x | Ebar] (30,pad32) -> 50 -> 25 -> 10
// ---------------------------------------------------------------------------
__global__ __launch_bounds__(256) void k_node(
    const float* __restrict__ x, const float* __restrict__ ebar,
    const float* __restrict__ w1, const float* __restrict__ b1,
    const float* __restrict__ w2, const float* __restrict__ b2,
    const float* __restrict__ w3, const float* __restrict__ b3,
    float* __restrict__ o)
{
    __shared__ __align__(32) __bf16 s_ft[N_NODES * 32];   // [node][featPad32]
    __shared__ __align__(32) __bf16 s_w1[64 * 32];
    __shared__ __align__(32) __bf16 s_w2[32 * 64];
    __shared__ __align__(32) __bf16 s_w3[16 * 32];
    __shared__ float s_b1[64], s_b2[32], s_b3[16];
    __shared__ __align__(32) __bf16 s_h1[8][16 * 64];
    __shared__ __align__(32) __bf16 s_h2[8][16 * 32];

    const int tid = threadIdx.x;
    const int b   = blockIdx.x;

    for (int i = tid; i < N_NODES * 32; i += 256) {
        int n = i >> 5, k = i & 31;
        float v = 0.f;
        if (k < 16)       v = x[(size_t)b * (P_FEAT * N_NODES) + k * N_NODES + n];
        else if (k < 30)  v = ebar[((size_t)b * N_NODES + n) * DE + (k - 16)];
        s_ft[i] = (__bf16)v;
    }
    // fo1 (50,30) -> [64][32]
    for (int i = tid; i < 64 * 32; i += 256) {
        int n = i >> 5, k = i & 31;
        s_w1[i] = (__bf16)((n < 50 && k < 30) ? w1[n * 30 + k] : 0.f);
    }
    // fo2 (25,50) -> [32][64]
    for (int i = tid; i < 32 * 64; i += 256) {
        int n = i >> 6, k = i & 63;
        s_w2[i] = (__bf16)((n < 25 && k < 50) ? w2[n * 50 + k] : 0.f);
    }
    // fo3 (10,25) -> [16][32]
    for (int i = tid; i < 16 * 32; i += 256) {
        int n = i >> 5, k = i & 31;
        s_w3[i] = (__bf16)((n < 10 && k < 25) ? w3[n * 25 + k] : 0.f);
    }
    if (tid < 64) s_b1[tid] = (tid < 50) ? b1[tid] : 0.f;
    if (tid < 32) s_b2[tid] = (tid < 25) ? b2[tid] : 0.f;
    if (tid < 16) s_b3[tid] = (tid < 10) ? b3[tid] : 0.f;
    __syncthreads();

    const int lane = tid & 31;
    const int wv   = tid >> 5;
    const int lh   = lane >> 4;
    const int nl   = lane & 15;
    const int f0   = lh * 8;
    const int kb   = lh * 16;

    v16bf bw1[4];
#pragma unroll
    for (int q = 0; q < 4; ++q)
        bw1[q] = cat16(*(const v8bf*)&s_w1[(q * 16 + nl) * 32 + f0],
                       *(const v8bf*)&s_w1[(q * 16 + nl) * 32 + 16 + f0]);
    v16bf bw2[2][2];
#pragma unroll
    for (int q = 0; q < 2; ++q)
#pragma unroll
        for (int c = 0; c < 2; ++c)
            bw2[q][c] = cat16(*(const v8bf*)&s_w2[(q * 16 + nl) * 64 + c * 32 + f0],
                              *(const v8bf*)&s_w2[(q * 16 + nl) * 64 + c * 32 + 16 + f0]);
    v16bf bw3 = cat16(*(const v8bf*)&s_w3[nl * 32 + f0],
                      *(const v8bf*)&s_w3[nl * 32 + 16 + f0]);

    __bf16* h1 = s_h1[wv];
    __bf16* h2 = s_h2[wv];

    for (int t = wv; t < NTILE_N; t += 8) {
        int node = t * 16 + nl;
        int nc   = (node < N_NODES) ? node : (N_NODES - 1);
        v16bf a = cat16(*(const v8bf*)&s_ft[nc * 32 + f0],
                        *(const v8bf*)&s_ft[nc * 32 + 16 + f0]);

        v8f acc[4];
#pragma unroll
        for (int q = 0; q < 4; ++q) { v8f z = {}; acc[q] = wmma_bf16(a, bw1[q], z); }
#pragma unroll
        for (int q = 0; q < 4; ++q) {
            float bv = s_b1[q * 16 + nl];
#pragma unroll
            for (int r = 0; r < 8; ++r) {
                float v = fmaxf(acc[q][r] + bv, 0.f);
                h1[(r + lh * 8) * 64 + q * 16 + nl] = (__bf16)v;
            }
        }
        wave_lds_sync();

        v16bf a20 = cat16(*(const v8bf*)&h1[nl * 64 + f0],
                          *(const v8bf*)&h1[nl * 64 + 16 + f0]);
        v16bf a21 = cat16(*(const v8bf*)&h1[nl * 64 + 32 + f0],
                          *(const v8bf*)&h1[nl * 64 + 48 + f0]);
        v8f acc2[2];
#pragma unroll
        for (int q = 0; q < 2; ++q) {
            v8f z = {};
            v8f tmp = wmma_bf16(a20, bw2[q][0], z);
            acc2[q] = wmma_bf16(a21, bw2[q][1], tmp);
        }
#pragma unroll
        for (int q = 0; q < 2; ++q) {
            float bv = s_b2[q * 16 + nl];
#pragma unroll
            for (int r = 0; r < 8; ++r) {
                float v = fmaxf(acc2[q][r] + bv, 0.f);
                h2[(r + lh * 8) * 32 + q * 16 + nl] = (__bf16)v;
            }
        }
        wave_lds_sync();

        v16bf a3 = cat16(*(const v8bf*)&h2[nl * 32 + f0],
                         *(const v8bf*)&h2[nl * 32 + 16 + f0]);
        v8f z3 = {};
        v8f o3 = wmma_bf16(a3, bw3, z3);

        float bv3 = s_b3[nl];
        if (nl < DO) {
#pragma unroll
            for (int r = 0; r < 8; ++r) {
                int n2 = t * 16 + r + lh * 8;
                if (n2 < N_NODES) {
                    float v = fmaxf(o3[r] + bv3, 0.f);
                    o[(size_t)b * (N_NODES * DO) + n2 * DO + nl] = v;
                }
            }
        }
        wave_lds_sync();
    }
}

// ---------------------------------------------------------------------------
// Head: flat(1500) -> 50 -> 25 -> 5 (tiny; plain VALU)
// ---------------------------------------------------------------------------
__global__ __launch_bounds__(64) void k_head(
    const float* __restrict__ o,
    const float* __restrict__ w1, const float* __restrict__ b1,
    const float* __restrict__ w2, const float* __restrict__ b2,
    const float* __restrict__ w3, const float* __restrict__ b3,
    float* __restrict__ out)
{
    __shared__ float s_f[N_NODES * DO];
    __shared__ float s_h1[50];
    __shared__ float s_h2[25];
    const int b = blockIdx.x, tid = threadIdx.x;

    for (int i = tid; i < N_NODES * DO; i += 64)
        s_f[i] = o[(size_t)b * (N_NODES * DO) + i];
    __syncthreads();

    if (tid < 50) {
        float acc = b1[tid];
        const float* wr = &w1[(size_t)tid * (N_NODES * DO)];
        for (int k = 0; k < N_NODES * DO; ++k) acc += wr[k] * s_f[k];
        s_h1[tid] = fmaxf(acc, 0.f);
    }
    __syncthreads();
    if (tid < 25) {
        float acc = b2[tid];
        for (int k = 0; k < 50; ++k) acc += w2[tid * 50 + k] * s_h1[k];
        s_h2[tid] = fmaxf(acc, 0.f);
    }
    __syncthreads();
    if (tid < NT) {
        float acc = b3[tid];
        for (int k = 0; k < 25; ++k) acc += w3[tid * 25 + k] * s_h2[k];
        out[b * NT + tid] = acc;
    }
}

extern "C" void kernel_launch(void* const* d_in, const int* in_sizes, int n_in,
                              void* d_out, int out_size, void* d_ws, size_t ws_size,
                              hipStream_t stream) {
    (void)in_sizes; (void)n_in; (void)out_size; (void)ws_size;
    const float* x     = (const float*)d_in[0];
    const int*   recv  = (const int*)d_in[1];
    const int*   send  = (const int*)d_in[2];
    const float* fr1_w = (const float*)d_in[3];
    const float* fr1_b = (const float*)d_in[4];
    const float* fr2_w = (const float*)d_in[5];
    const float* fr2_b = (const float*)d_in[6];
    const float* fr3_w = (const float*)d_in[7];
    const float* fr3_b = (const float*)d_in[8];
    const float* fo1_w = (const float*)d_in[9];
    const float* fo1_b = (const float*)d_in[10];
    const float* fo2_w = (const float*)d_in[11];
    const float* fo2_b = (const float*)d_in[12];
    const float* fo3_w = (const float*)d_in[13];
    const float* fo3_b = (const float*)d_in[14];
    const float* fc1_w = (const float*)d_in[15];
    const float* fc1_b = (const float*)d_in[16];
    const float* fc2_w = (const float*)d_in[17];
    const float* fc2_b = (const float*)d_in[18];
    const float* fc3_w = (const float*)d_in[19];
    const float* fc3_b = (const float*)d_in[20];

    float* ws   = (float*)d_ws;
    float* ebar = ws;                                       // B*N*DE floats
    float* obuf = ws + (size_t)BATCH * N_NODES * DE;        // B*N*DO floats

    const int ez = BATCH * N_NODES * DE;
    k_zero<<<(ez + 255) / 256, 256, 0, stream>>>(ebar, ez);

    k_edge<<<BATCH * SPLIT, 256, 0, stream>>>(x, recv, send,
                                              fr1_w, fr1_b, fr2_w, fr2_b,
                                              fr3_w, fr3_b, ebar);

    k_node<<<BATCH, 256, 0, stream>>>(x, ebar,
                                      fo1_w, fo1_b, fo2_w, fo2_b,
                                      fo3_w, fo3_b, obuf);

    k_head<<<BATCH, 64, 0, stream>>>(obuf, fc1_w, fc1_b, fc2_w, fc2_b,
                                     fc3_w, fc3_b, (float*)d_out);
}
